// MLA_20873541059075
// MI455X (gfx1250) — matrix operand order repaired and checked by reading, not compiled, not run on hardware
//
#include <hip/hip_runtime.h>
#include <hip/hip_bf16.h>
#include <stdint.h>

#define DIM     2048
#define H       16
#define KV_RANK 512
#define NOPE    128
#define ROPE_D  64
#define VD      128
#define QKD     192           // NOPE + ROPE_D
#define BATCH   2
#define SEQ     2048
#define NTOK    (BATCH*SEQ)   // 4096
#define KVA_N   (KV_RANK + ROPE_D)     // 576
#define KVB_N   (H*(NOPE+VD))          // 4096
#define EPS_LN  1e-5f
#define ATT_SCALE 0.07216878364870323f // 192^-0.5

typedef __bf16 bf16;
typedef __attribute__((ext_vector_type(16))) __bf16 v16bf;
typedef __attribute__((ext_vector_type(8)))  float  v8f;

union AB16 { v16bf v; uint4 q[2]; };

// ---------------------------------------------------------------------------
// WMMA helpers (CDNA5 wave32, D = A(16x32) * B(32x16) + C, f32 accum)
// ---------------------------------------------------------------------------
__device__ inline v8f wmma_bf16(v16bf a, v16bf b, v8f c) {
  return __builtin_amdgcn_wmma_f32_16x16x32_bf16(
      /*neg_a=*/false, a, /*neg_b=*/false, b,
      /*c_mod=*/(short)0, c, /*reuse_a=*/false, /*reuse_b=*/false);
}

// A-matrix 16x32 fragment from a row-major matrix (tile origin at `base`).
// ISA layout: lanes 0-15 -> rows 0-15 with K {0..7, 16..23};
//             lanes 16-31 -> rows 0-15 with K {8..15, 24..31}.
__device__ inline v16bf load_frag_a(const bf16* base, int ld, int lane) {
  const bf16* p = base + (size_t)(lane & 15) * (size_t)ld + ((lane >> 4) << 3);
  AB16 f;
  f.q[0] = *(const uint4*)(p);
  f.q[1] = *(const uint4*)(p + 16);
  return f.v;
}

// B-matrix 32x16 fragment where the source is stored "N-major" (each B column
// n is a contiguous K-run, i.e. a row of W(N,K)).  Tile origin (n0,k0) at base.
// ISA layout: lanes 0-15 -> K {0..7, 16..23} of column lane; lanes 16-31 -> K {8..15, 24..31}.
__device__ inline v16bf load_frag_b(const bf16* base, int ld, int lane) {
  const bf16* p = base + (size_t)(lane & 15) * (size_t)ld + ((lane >> 4) << 3);
  AB16 f;
  f.q[0] = *(const uint4*)(p);
  f.q[1] = *(const uint4*)(p + 16);
  return f.v;
}

// ---------------------------------------------------------------------------
// Generic GEMM: C[M,N] (f32) = A[M,K] (bf16) * W[N,K]^T (bf16)
// One wave computes a 64x64 C tile via 4x4 WMMA frags (high operand reuse:
// 8 fragment loads feed 16 WMMAs per K-step -> ~32 FLOP/byte from cache).
// ---------------------------------------------------------------------------
__global__ void __launch_bounds__(256)
gemm_xt_bf16(const bf16* __restrict__ A, const bf16* __restrict__ W,
             float* __restrict__ C, int M, int N, int K) {
  int gwave = (int)((blockIdx.x * 256u + threadIdx.x) >> 5);
  int lane  = threadIdx.x & 31;
  int ntn   = N >> 6;
  int tm    = gwave / ntn;
  int tn    = gwave % ntn;
  if (tm * 64 >= M) return;

  const bf16* abase[4];
  const bf16* bbase[4];
  for (int i = 0; i < 4; ++i) {
    abase[i] = A + (size_t)(tm * 64 + i * 16) * (size_t)K;
    bbase[i] = W + (size_t)(tn * 64 + i * 16) * (size_t)K;
  }

  v8f acc[4][4] = {};
  for (int k0 = 0; k0 < K; k0 += 32) {
    v16bf af[4], bfb[4];
    for (int i = 0; i < 4; ++i) af[i]  = load_frag_a(abase[i] + k0, K, lane);
    for (int j = 0; j < 4; ++j) bfb[j] = load_frag_b(bbase[j] + k0, K, lane);
    for (int i = 0; i < 4; ++i)
      for (int j = 0; j < 4; ++j)
        acc[i][j] = wmma_bf16(af[i], bfb[j], acc[i][j]);
  }

  // C layout: VGPR i -> row (lane<16 ? i : 8+i), col = lane%16
  int rlo = (lane < 16) ? 0 : 8;
  int cl  = lane & 15;
  for (int mi = 0; mi < 4; ++mi)
    for (int ni = 0; ni < 4; ++ni)
      for (int i = 0; i < 8; ++i) {
        size_t r = (size_t)(tm * 64 + mi * 16 + rlo + i);
        C[r * N + tn * 64 + ni * 16 + cl] = acc[mi][ni][i];
      }
}

// ---------------------------------------------------------------------------
// f32 -> bf16 convert
// ---------------------------------------------------------------------------
__global__ void __launch_bounds__(256)
f32_to_bf16_kernel(const float* __restrict__ src, bf16* __restrict__ dst, int n) {
  int i = (int)(blockIdx.x * 256u + threadIdx.x);
  if (i < n) dst[i] = (bf16)src[i];
}

// ---------------------------------------------------------------------------
// q prep: copy nope part, RoPE the pe part.  qraw[tok, H*QKD] -> qfull bf16.
// ---------------------------------------------------------------------------
__global__ void __launch_bounds__(256)
qprep_kernel(const float* __restrict__ qraw, bf16* __restrict__ qfull,
             const int* __restrict__ start_pos) {
  int tok = blockIdx.x;
  int s   = tok % SEQ;
  float pos = (float)(s + start_pos[0]);
  const float* src = qraw  + (size_t)tok * (H * QKD);
  bf16*        dst = qfull + (size_t)tok * (H * QKD);

  for (int idx = threadIdx.x; idx < H * NOPE; idx += 256) {
    int h = idx >> 7, d = idx & 127;
    dst[h * QKD + d] = (bf16)src[h * QKD + d];
  }
  const float lf = logf(10000.f) * (2.f / ROPE_D);
  for (int idx = threadIdx.x; idx < H * (ROPE_D / 2); idx += 256) {
    int h = idx >> 5, i = idx & 31;
    float ang = pos * __expf(-lf * (float)i);
    float sn, cs; __sincosf(ang, &sn, &cs);
    float x1 = src[h * QKD + NOPE + i];
    float x2 = src[h * QKD + NOPE + 32 + i];
    dst[h * QKD + NOPE + i]      = (bf16)(x1 * cs - x2 * sn);
    dst[h * QKD + NOPE + 32 + i] = (bf16)(x2 * cs + x1 * sn);
  }
}

// ---------------------------------------------------------------------------
// kv prep: layernorm the latent (512) -> kvn bf16; RoPE k_pe (64) -> kpe bf16.
// One block per token.
// ---------------------------------------------------------------------------
__global__ void __launch_bounds__(256)
kvprep_kernel(const float* __restrict__ kvall, const float* __restrict__ lnw,
              const float* __restrict__ lnb, bf16* __restrict__ kvn,
              bf16* __restrict__ kpe, const int* __restrict__ start_pos) {
  __shared__ float red[256];
  int tok = blockIdx.x;
  int tid = threadIdx.x;
  const float* src = kvall + (size_t)tok * KVA_N;

  float sum = 0.f;
  for (int i = tid; i < KV_RANK; i += 256) sum += src[i];
  red[tid] = sum; __syncthreads();
  for (int off = 128; off > 0; off >>= 1) {
    if (tid < off) red[tid] += red[tid + off];
    __syncthreads();
  }
  float mu = red[0] * (1.f / KV_RANK);
  __syncthreads();

  float vs = 0.f;
  for (int i = tid; i < KV_RANK; i += 256) { float d = src[i] - mu; vs += d * d; }
  red[tid] = vs; __syncthreads();
  for (int off = 128; off > 0; off >>= 1) {
    if (tid < off) red[tid] += red[tid + off];
    __syncthreads();
  }
  float rstd = rsqrtf(red[0] * (1.f / KV_RANK) + EPS_LN);

  for (int i = tid; i < KV_RANK; i += 256)
    kvn[(size_t)tok * KV_RANK + i] = (bf16)((src[i] - mu) * rstd * lnw[i] + lnb[i]);

  int s = tok % SEQ;
  float pos = (float)(s + start_pos[0]);
  const float lf = logf(10000.f) * (2.f / ROPE_D);
  for (int i = tid; i < ROPE_D / 2; i += 256) {
    float ang = pos * __expf(-lf * (float)i);
    float sn, cs; __sincosf(ang, &sn, &cs);
    float x1 = src[KV_RANK + i];
    float x2 = src[KV_RANK + 32 + i];
    kpe[(size_t)tok * ROPE_D + i]      = (bf16)(x1 * cs - x2 * sn);
    kpe[(size_t)tok * ROPE_D + 32 + i] = (bf16)(x2 * cs + x1 * sn);
  }
}

// ---------------------------------------------------------------------------
// k/v prep: split kvb GEMM output into k_full[B,S,H,QKD] (with broadcast k_pe
// tail) and pre-transposed vT[B,H,VD,SEQ] (bf16) for contiguous P*V B-frags.
// ---------------------------------------------------------------------------
__global__ void __launch_bounds__(256)
kprep_kernel(const float* __restrict__ kvbout, const bf16* __restrict__ kpe,
             bf16* __restrict__ kfull, bf16* __restrict__ vT) {
  int tok = blockIdx.x;
  int b = tok / SEQ, s = tok % SEQ;
  const float* src = kvbout + (size_t)tok * KVB_N;
  for (int f = threadIdx.x; f < KVB_N; f += 256) {
    int h = f >> 8;       // / (NOPE+VD)
    int d = f & 255;
    float val = src[f];
    if (d < NOPE) {
      kfull[(((size_t)b * SEQ + s) * H + h) * QKD + d] = (bf16)val;
    } else {
      vT[(((size_t)b * H + h) * VD + (d - NOPE)) * SEQ + s] = (bf16)val;
    }
  }
  for (int j = threadIdx.x; j < H * ROPE_D; j += 256) {
    int h = j >> 6, i = j & 63;
    kfull[(((size_t)b * SEQ + s) * H + h) * QKD + NOPE + i] =
        kpe[(size_t)tok * ROPE_D + i];
  }
}

// ---------------------------------------------------------------------------
// Flash attention, causal.  Each wave owns 16 query rows of one (b,h);
// block = 8 waves = 128 query rows.  Key tiles of 32 with online softmax.
// ---------------------------------------------------------------------------
__global__ void __launch_bounds__(256)
mla_attn_kernel(const bf16* __restrict__ qfull, const bf16* __restrict__ kfull,
                const bf16* __restrict__ vT, bf16* __restrict__ attn_out) {
  __shared__ __align__(16) bf16 pbuf[8][16 * 32];
  int wv   = threadIdx.x >> 5;
  int lane = threadIdx.x & 31;

  int qblocks = SEQ / 128;
  int qb = blockIdx.x % qblocks;
  int bh = blockIdx.x / qblocks;
  int b = bh / H, h = bh % H;
  int qrow0 = qb * 128 + wv * 16;
  const int ldq = H * QKD;    // 3072

  const bf16* qbase = qfull + ((size_t)(b * SEQ) * H + h) * QKD +
                      (size_t)qrow0 * ldq;
  const bf16* kbase = kfull + ((size_t)(b * SEQ) * H + h) * QKD;
  const bf16* vbase = vT + ((size_t)(b * H + h) * VD) * SEQ;

  // Q fragments: 16 rows x 192 = 6 K-chunks, held in registers for whole loop
  v16bf qf[6];
  for (int c = 0; c < 6; ++c) qf[c] = load_frag_a(qbase + c * 32, ldq, lane);

  v8f o[8] = {};                       // 16 x 128 f32 accumulator
  float m[8], l[8];
  for (int i = 0; i < 8; ++i) { m[i] = -1e30f; l[i] = 0.f; }

  int rlo = (lane < 16) ? 0 : 8;
  int cl  = lane & 15;

  int ktiles = (qrow0 + 15) / 32 + 1;  // causal: keys <= qrow0+15
  for (int kt = 0; kt < ktiles; ++kt) {
    int key0 = kt * 32;
    const bf16* kb0 = kbase + (size_t)key0 * ldq;
    const bf16* kb1 = kbase + (size_t)(key0 + 16) * ldq;

    v8f s0 = {}, s1 = {};
    for (int c = 0; c < 6; ++c) {
      s0 = wmma_bf16(qf[c], load_frag_b(kb0 + c * 32, ldq, lane), s0);
      s1 = wmma_bf16(qf[c], load_frag_b(kb1 + c * 32, ldq, lane), s1);
    }

    // online softmax per row slot (row = qrow0 + rlo + i, col = key0 + cl)
    float p0[8], p1[8];
    for (int i = 0; i < 8; ++i) {
      int rg = qrow0 + rlo + i;
      float v0 = s0[i] * ATT_SCALE;
      float v1 = s1[i] * ATT_SCALE;
      if (key0 + cl > rg)      v0 = -1e30f;
      if (key0 + 16 + cl > rg) v1 = -1e30f;
      float mx = fmaxf(v0, v1);
      for (int msk = 1; msk <= 8; msk <<= 1)
        mx = fmaxf(mx, __shfl_xor(mx, msk, 32));
      float mnew  = fmaxf(m[i], mx);
      float alpha = __expf(m[i] - mnew);
      float e0 = __expf(v0 - mnew);
      float e1 = __expf(v1 - mnew);
      float rs = e0 + e1;
      for (int msk = 1; msk <= 8; msk <<= 1)
        rs += __shfl_xor(rs, msk, 32);
      l[i] = l[i] * alpha + rs;
      m[i] = mnew;
      p0[i] = e0; p1[i] = e1;
      for (int j = 0; j < 8; ++j) o[j][i] *= alpha;
    }

    // re-layout P (C layout f32) -> A fragment (bf16) through per-wave LDS;
    // same-wave DS ops are in-order, no barrier required.
    for (int i = 0; i < 8; ++i) {
      int r = rlo + i;
      pbuf[wv][r * 32 + cl]      = (bf16)p0[i];
      pbuf[wv][r * 32 + 16 + cl] = (bf16)p1[i];
    }
    const bf16* pp = &pbuf[wv][0];
    int prow = lane & 15, pkg = lane >> 4;
    AB16 pf;
    pf.q[0] = *(const uint4*)(pp + prow * 32 + pkg * 8);
    pf.q[1] = *(const uint4*)(pp + prow * 32 + pkg * 8 + 16);
    v16bf pa = pf.v;

    for (int j = 0; j < 8; ++j) {
      const bf16* vb = vbase + (size_t)(j * 16) * SEQ + key0;
      o[j] = wmma_bf16(pa, load_frag_b(vb, SEQ, lane), o[j]);
    }
  }

  // normalize and store bf16 to attn_out[B,S,H,VD]
  for (int i = 0; i < 8; ++i) {
    float inv = 1.f / l[i];
    int sg = qrow0 + rlo + i;
    bf16* dst = attn_out + (((size_t)(b * SEQ + sg)) * H + h) * VD;
    for (int j = 0; j < 8; ++j)
      dst[j * 16 + cl] = (bf16)(o[j][i] * inv);
  }
}

// ---------------------------------------------------------------------------
// Host launcher
// ---------------------------------------------------------------------------
extern "C" void kernel_launch(void* const* d_in, const int* in_sizes, int n_in,
                              void* d_out, int out_size, void* d_ws, size_t ws_size,
                              hipStream_t stream) {
  const float* x     = (const float*)d_in[0];
  const float* wq    = (const float*)d_in[1];
  const float* wkva  = (const float*)d_in[2];
  const float* lnw   = (const float*)d_in[3];
  const float* lnb   = (const float*)d_in[4];
  const float* wkvb  = (const float*)d_in[5];
  const float* wo    = (const float*)d_in[6];
  const int*   spos  = (const int*)d_in[7];
  float* out = (float*)d_out;

  // workspace carving (256B aligned)
  size_t off = 0;
  char* base = (char*)d_ws;
  auto alloc = [&](size_t bytes) -> void* {
    void* p = base + off;
    off += (bytes + 255) & ~(size_t)255;
    return p;
  };
  bf16*  x_bf    = (bf16*) alloc((size_t)NTOK * DIM * 2);
  bf16*  wq_bf   = (bf16*) alloc((size_t)H * QKD * DIM * 2);
  bf16*  wkva_bf = (bf16*) alloc((size_t)KVA_N * DIM * 2);
  bf16*  wkvb_bf = (bf16*) alloc((size_t)KVB_N * KV_RANK * 2);
  bf16*  wo_bf   = (bf16*) alloc((size_t)DIM * (H * VD) * 2);
  float* qraw    = (float*)alloc((size_t)NTOK * H * QKD * 4);
  float* kvall   = (float*)alloc((size_t)NTOK * KVA_N * 4);
  bf16*  qfull   = (bf16*) alloc((size_t)NTOK * H * QKD * 2);
  bf16*  kvn     = (bf16*) alloc((size_t)NTOK * KV_RANK * 2);
  bf16*  kpe     = (bf16*) alloc((size_t)NTOK * ROPE_D * 2);
  float* kvbout  = (float*)alloc((size_t)NTOK * KVB_N * 4);
  bf16*  kfull   = (bf16*) alloc((size_t)NTOK * H * QKD * 2);
  bf16*  vTb     = (bf16*) alloc((size_t)BATCH * H * VD * SEQ * 2);
  bf16*  attnb   = (bf16*) alloc((size_t)NTOK * H * VD * 2);

  auto cdiv = [](long a, long b) { return (int)((a + b - 1) / b); };
  auto cvt = [&](const float* s, bf16* d, long n) {
    f32_to_bf16_kernel<<<cdiv(n, 256), 256, 0, stream>>>(s, d, (int)n);
  };
  // 1) converts
  cvt(x,    x_bf,    (long)NTOK * DIM);
  cvt(wq,   wq_bf,   (long)H * QKD * DIM);
  cvt(wkva, wkva_bf, (long)KVA_N * DIM);
  cvt(wkvb, wkvb_bf, (long)KVB_N * KV_RANK);
  cvt(wo,   wo_bf,   (long)DIM * H * VD);

  auto gemm = [&](const bf16* A, const bf16* W, float* C, int M, int N, int K) {
    long waves = (long)(M / 64) * (N / 64);
    gemm_xt_bf16<<<cdiv(waves, 8), 256, 0, stream>>>(A, W, C, M, N, K);
  };
  // 2) projections
  gemm(x_bf, wq_bf,   qraw,  NTOK, H * QKD, DIM);    // q = x @ wq^T
  gemm(x_bf, wkva_bf, kvall, NTOK, KVA_N,   DIM);    // kv_all = x @ wkv_a^T

  // 3) elementwise prep
  qprep_kernel <<<NTOK, 256, 0, stream>>>(qraw, qfull, spos);
  kvprep_kernel<<<NTOK, 256, 0, stream>>>(kvall, lnw, lnb, kvn, kpe, spos);

  // 4) k_nope / v projection from latent
  gemm(kvn, wkvb_bf, kvbout, NTOK, KVB_N, KV_RANK);
  kprep_kernel<<<NTOK, 256, 0, stream>>>(kvbout, kpe, kfull, vTb);

  // 5) attention (flash, causal)
  mla_attn_kernel<<<BATCH * H * (SEQ / 128), 256, 0, stream>>>(qfull, kfull, vTb, attnb);

  // 6) output projection straight into d_out (f32)
  gemm(attnb, wo_bf, out, NTOK, DIM, H * VD);
}
